// LlamaAttention_35811437314523
// MI455X (gfx1250) — matrix operand order repaired and checked by reading, not compiled, not run on hardware
//
#include <hip/hip_runtime.h>
#include <math.h>

#define B_    4
#define S_    2048
#define DIM_  2048
#define HEADS_ 16
#define HD_   128

typedef __attribute__((ext_vector_type(16))) __bf16 v16bf;
typedef __attribute__((ext_vector_type(8)))  __bf16 v8bf;
typedef __attribute__((ext_vector_type(8)))  float  v8f;

__device__ __forceinline__ v16bf make_frag(v8bf lo, v8bf hi) {
  v16bf r;
#pragma unroll
  for (int i = 0; i < 8; ++i) { r[i] = lo[i]; r[8 + i] = hi[i]; }
  return r;
}

// A fragment 16(M)x32(K), source row-major [M][ldk], `a` = tile origin.
// lane: m = lane&15 ; halves 0..7 -> K = (lane>>4)*8 + h ; halves 8..15 -> K = 16 + (lane>>4)*8 + h
__device__ __forceinline__ v16bf load_a16x32(const __bf16* a, int ldk, int lane) {
  const int m  = lane & 15;
  const int kh = (lane >> 4) << 3;
  const __bf16* row = a + m * ldk;
  v8bf lo = *(const v8bf*)(row + kh);
  v8bf hi = *(const v8bf*)(row + 16 + kh);
  return make_frag(lo, hi);
}

// B fragment 32(K)x16(N) loaded from B^T row-major [N][ldk], `bt` = tile origin.
// lane: n = lane&15 ; halves 0..7 -> K = (lane>>4)*8 + h ; halves 8..15 -> K = 16 + (lane>>4)*8 + h
__device__ __forceinline__ v16bf load_bt16x32(const __bf16* bt, int ldk, int lane) {
  const int n  = lane & 15;
  const int kb = (lane >> 4) << 3;
  const __bf16* row = bt + n * ldk + kb;
  v8bf lo = *(const v8bf*)(row);
  v8bf hi = *(const v8bf*)(row + 16);
  return make_frag(lo, hi);
}

__device__ __forceinline__ v8f wmma_bf16(v16bf a, v16bf b, v8f c) {
  return __builtin_amdgcn_wmma_f32_16x16x32_bf16(false, a, false, b, (short)0, c, false, false);
}

__device__ __forceinline__ v8f vzero8() {
  v8f z = {0.f, 0.f, 0.f, 0.f, 0.f, 0.f, 0.f, 0.f};
  return z;
}

// ---------------- elementwise convert fp32 -> bf16 ----------------
__global__ void convert_f32_to_bf16(const float* __restrict__ in,
                                    __bf16* __restrict__ out, int n) {
  int i = blockIdx.x * blockDim.x + threadIdx.x;
  if (i < n) out[i] = (__bf16)in[i];
}

// ---------------- 2048x2048 transpose + convert (LDS tiled) ----------------
// out[n][k] = (bf16) in[k][n]
__global__ __launch_bounds__(256) void transpose_to_bf16(const float* __restrict__ in,
                                                         __bf16* __restrict__ out) {
  __shared__ float tile[32][33];
  const int k0 = blockIdx.x * 32, n0 = blockIdx.y * 32;
  const int tx = threadIdx.x & 31, ty = threadIdx.x >> 5;
#pragma unroll
  for (int r = ty; r < 32; r += 8)
    tile[r][tx] = in[(size_t)(k0 + r) * DIM_ + (n0 + tx)];
  __syncthreads();
#pragma unroll
  for (int r = ty; r < 32; r += 8)
    out[(size_t)(n0 + r) * DIM_ + (k0 + tx)] = (__bf16)tile[tx][r];
}

// ---------------- fused QKV GEMM + bias + RoPE ----------------
// Wave tile: 16(M) x 128(N) == one full head of N columns. z selects q/k/v.
__global__ __launch_bounds__(256) void qkv_rope_gemm(
    const __bf16* __restrict__ xb,
    const __bf16* __restrict__ wqt, const __bf16* __restrict__ wkt,
    const __bf16* __restrict__ wvt,
    const float* __restrict__ bq, const float* __restrict__ bk,
    const float* __restrict__ bv,
    __bf16* __restrict__ qr, __bf16* __restrict__ kr, __bf16* __restrict__ vt) {
  const int lane = threadIdx.x & 31, wave = threadIdx.x >> 5;
  const int z = blockIdx.z;
  const int h = blockIdx.y;
  const int m0 = (blockIdx.x * 8 + wave) * 16;

  const __bf16* Wt  = (z == 0) ? wqt : (z == 1) ? wkt : wvt;
  const float* bias = (z == 0) ? bq  : (z == 1) ? bk  : bv;

  v8f acc[8];
#pragma unroll
  for (int t = 0; t < 8; ++t) acc[t] = vzero8();

  const __bf16* Abase = xb + (size_t)m0 * DIM_;
  for (int k0 = 0; k0 < DIM_; k0 += 32) {
    v16bf a = load_a16x32(Abase + k0, DIM_, lane);
#pragma unroll
    for (int t = 0; t < 8; ++t) {
      v16bf b = load_bt16x32(Wt + (size_t)(h * HD_ + t * 16) * DIM_ + k0, DIM_, lane);
      acc[t] = wmma_bf16(a, b, acc[t]);
    }
  }

  const int n = lane & 15, rs = lane >> 4;
  float bcol[8];
#pragma unroll
  for (int t = 0; t < 8; ++t) bcol[t] = bias[h * HD_ + t * 16 + n];

  if (z < 2) {
    // RoPE: pair (i, i+64) within the head; i = t*16 + n for t in 0..3
    const float sc = (z == 0) ? 0.088388347648318447f /* 1/sqrt(128) */ : 1.0f;
    float fr[4];
#pragma unroll
    for (int t = 0; t < 4; ++t) {
      float i = (float)(t * 16 + n);
      fr[t] = __powf(10000.0f, -i * (1.0f / 64.0f));
    }
    __bf16* dst = (z == 0) ? qr : kr;
#pragma unroll
    for (int r = 0; r < 8; ++r) {
      const int mg = m0 + r + 8 * rs;
      const int bidx = mg / S_, s = mg % S_;
      __bf16* drow = dst + (((size_t)bidx * HEADS_ + h) * S_ + s) * HD_;
      const float sf = (float)s;
#pragma unroll
      for (int t = 0; t < 4; ++t) {
        float re = acc[t][r] + bcol[t];
        float im = acc[t + 4][r] + bcol[t + 4];
        float sn, cs;
        __sincosf(sf * fr[t], &sn, &cs);
        drow[t * 16 + n]      = (__bf16)((re * cs - im * sn) * sc);
        drow[64 + t * 16 + n] = (__bf16)((im * cs + re * sn) * sc);
      }
    }
  } else {
    // V: store transposed [B][H][HD][S] for B^T-style fragments in P·V
#pragma unroll
    for (int r = 0; r < 8; ++r) {
      const int mg = m0 + r + 8 * rs;
      const int bidx = mg / S_, s = mg % S_;
      __bf16* dcol = vt + (((size_t)bidx * HEADS_ + h) * HD_) * (size_t)S_;
#pragma unroll
      for (int t = 0; t < 8; ++t)
        dcol[(size_t)(t * 16 + n) * S_ + s] = (__bf16)(acc[t][r] + bcol[t]);
    }
  }
}

// ---------------- flash attention: one wave per 16-row query tile ----------------
__global__ __launch_bounds__(256) void flash_attn(
    const __bf16* __restrict__ qr, const __bf16* __restrict__ kr,
    const __bf16* __restrict__ vt, __bf16* __restrict__ attnout) {
  __shared__ __align__(16) __bf16 plds[8][16][32];  // per-wave P staging (C->A relayout)
  const int lane = threadIdx.x & 31, wave = threadIdx.x >> 5;
  const int bh = blockIdx.y;
  const int bidx = bh >> 4, h = bh & 15;
  const int s0 = (blockIdx.x * 8 + wave) * 16;

  const __bf16* Q  = qr + ((size_t)bh * S_ + s0) * HD_;
  const __bf16* Kb = kr + (size_t)bh * S_ * HD_;
  const __bf16* Vb = vt + (size_t)bh * HD_ * (size_t)S_;

  v16bf qf[4];
#pragma unroll
  for (int kk = 0; kk < 4; ++kk) qf[kk] = load_a16x32(Q + kk * 32, HD_, lane);

  v8f o[8];
#pragma unroll
  for (int t = 0; t < 8; ++t) o[t] = vzero8();
  float mrow[8], lrow[8];
#pragma unroll
  for (int r = 0; r < 8; ++r) { mrow[r] = -1e30f; lrow[r] = 0.0f; }

  const int n = lane & 15, rs = lane >> 4;
  const __bf16* pbuf = &plds[wave][0][0];

  for (int kb = 0; kb < s0 + 16; kb += 32) {
    v8f sc0 = vzero8(), sc1 = vzero8();
#pragma unroll
    for (int kk = 0; kk < 4; ++kk) {
      v16bf b0 = load_bt16x32(Kb + (size_t)kb * HD_ + kk * 32, HD_, lane);
      sc0 = wmma_bf16(qf[kk], b0, sc0);
      v16bf b1 = load_bt16x32(Kb + (size_t)(kb + 16) * HD_ + kk * 32, HD_, lane);
      sc1 = wmma_bf16(qf[kk], b1, sc1);
    }
    const int sk0 = kb + n, sk1 = kb + 16 + n;
#pragma unroll
    for (int r = 0; r < 8; ++r) {
      const int sq = s0 + r + 8 * rs;
      float v0 = (sk0 <= sq) ? sc0[r] : -1e30f;   // causal mask
      float v1 = (sk1 <= sq) ? sc1[r] : -1e30f;
      float mx = fmaxf(v0, v1);
#pragma unroll
      for (int d = 1; d < 16; d <<= 1) mx = fmaxf(mx, __shfl_xor(mx, d, 32));
      const float mnew  = fmaxf(mrow[r], mx);
      const float scale = __expf(mrow[r] - mnew);
      const float p0 = __expf(v0 - mnew);
      const float p1 = __expf(v1 - mnew);
      float ps = p0 + p1;
#pragma unroll
      for (int d = 1; d < 16; d <<= 1) ps += __shfl_xor(ps, d, 32);
      lrow[r] = lrow[r] * scale + ps;
      mrow[r] = mnew;
#pragma unroll
      for (int t = 0; t < 8; ++t) o[t][r] = o[t][r] * scale;
      const int m = r + 8 * rs;
      plds[wave][m][n]      = (__bf16)p0;   // same-wave DS ops stay in order: no barrier
      plds[wave][m][16 + n] = (__bf16)p1;
    }
    v16bf pf = load_a16x32(pbuf, 32, lane);
#pragma unroll
    for (int t = 0; t < 8; ++t) {
      v16bf vb = load_bt16x32(Vb + (size_t)(t * 16) * S_ + kb, S_, lane);
      o[t] = wmma_bf16(pf, vb, o[t]);
    }
  }

#pragma unroll
  for (int r = 0; r < 8; ++r) {
    const float inv = 1.0f / lrow[r];
    const size_t row = ((size_t)bidx * S_ + s0 + r + 8 * rs) * DIM_ + h * HD_;
#pragma unroll
    for (int t = 0; t < 8; ++t)
      attnout[row + t * 16 + n] = (__bf16)(o[t][r] * inv);
  }
}

// ---------------- output projection + bias ----------------
__global__ __launch_bounds__(256) void out_proj_gemm(
    const __bf16* __restrict__ attn, const __bf16* __restrict__ wot,
    const float* __restrict__ bo, float* __restrict__ out) {
  const int lane = threadIdx.x & 31, wave = threadIdx.x >> 5;
  const int m0 = (blockIdx.x * 8 + wave) * 16;
  const int n0 = blockIdx.y * 128;

  v8f acc[8];
#pragma unroll
  for (int t = 0; t < 8; ++t) acc[t] = vzero8();

  const __bf16* Abase = attn + (size_t)m0 * DIM_;
  for (int k0 = 0; k0 < DIM_; k0 += 32) {
    v16bf a = load_a16x32(Abase + k0, DIM_, lane);
#pragma unroll
    for (int t = 0; t < 8; ++t) {
      v16bf b = load_bt16x32(wot + (size_t)(n0 + t * 16) * DIM_ + k0, DIM_, lane);
      acc[t] = wmma_bf16(a, b, acc[t]);
    }
  }
  const int n = lane & 15, rs = lane >> 4;
#pragma unroll
  for (int r = 0; r < 8; ++r) {
    const size_t row = (size_t)(m0 + r + 8 * rs) * DIM_;
#pragma unroll
    for (int t = 0; t < 8; ++t)
      out[row + n0 + t * 16 + n] = acc[t][r] + bo[n0 + t * 16 + n];
  }
}

extern "C" void kernel_launch(void* const* d_in, const int* in_sizes, int n_in,
                              void* d_out, int out_size, void* d_ws, size_t ws_size,
                              hipStream_t stream) {
  (void)in_sizes; (void)n_in; (void)out_size; (void)ws_size;
  const float* x  = (const float*)d_in[0];
  const float* wq = (const float*)d_in[1];
  const float* bq = (const float*)d_in[2];
  const float* wk = (const float*)d_in[3];
  const float* bk = (const float*)d_in[4];
  const float* wv = (const float*)d_in[5];
  const float* bv = (const float*)d_in[6];
  const float* wo = (const float*)d_in[7];
  const float* bo = (const float*)d_in[8];
  float* out = (float*)d_out;

  const size_t nBS = (size_t)B_ * S_ * DIM_;  // 16.7M elems
  const size_t nW  = (size_t)DIM_ * DIM_;     // 4.2M elems
  size_t off = 0;
  char* base = (char*)d_ws;
  auto alloc = [&](size_t bytes) { char* p = base + off; off += bytes; return p; };
  __bf16* xb  = (__bf16*)alloc(nBS * 2);
  __bf16* wqt = (__bf16*)alloc(nW * 2);
  __bf16* wkt = (__bf16*)alloc(nW * 2);
  __bf16* wvt = (__bf16*)alloc(nW * 2);
  __bf16* wot = (__bf16*)alloc(nW * 2);
  __bf16* qrb = (__bf16*)alloc(nBS * 2);
  __bf16* krb = (__bf16*)alloc(nBS * 2);
  __bf16* vtb = (__bf16*)alloc(nBS * 2);
  __bf16* atb = (__bf16*)alloc(nBS * 2);   // total ~192 MB

  convert_f32_to_bf16<<<(int)(nBS / 256), 256, 0, stream>>>(x, xb, (int)nBS);

  dim3 tg(DIM_ / 32, DIM_ / 32);
  transpose_to_bf16<<<tg, 256, 0, stream>>>(wq, wqt);
  transpose_to_bf16<<<tg, 256, 0, stream>>>(wk, wkt);
  transpose_to_bf16<<<tg, 256, 0, stream>>>(wv, wvt);
  transpose_to_bf16<<<tg, 256, 0, stream>>>(wo, wot);

  // M-tiles = 8192/16 = 512 -> 64 blocks of 8 waves; 16 heads; 3 matrices
  qkv_rope_gemm<<<dim3(64, 16, 3), 256, 0, stream>>>(xb, wqt, wkt, wvt,
                                                     bq, bk, bv, qrb, krb, vtb);

  // q-tiles = 2048/16 = 128 -> 16 blocks of 8 waves; 64 (b,h) pairs
  flash_attn<<<dim3(16, 64), 256, 0, stream>>>(qrb, krb, vtb, atb);

  out_proj_gemm<<<dim3(64, 16), 256, 0, stream>>>(atb, wot, bo, out);
}
